// GCN_24472723652990
// MI455X (gfx1250) — matrix operand order repaired and checked by reading, not compiled, not run on hardware
//
#include <hip/hip_runtime.h>
#include <hip/hip_bf16.h>

// ---------------------------------------------------------------------------
// GCN (3x GCNConv + BN + ReLU, then FC) for MI455X / gfx1250.
// Dense GEMMs: v_wmma_f32_16x16x32_bf16, register-blocked 64x32 per wave
// (4x2 tiles of 16x16 -> 8 WMMAs per 32-K step, 6 fragment loads).
// Sparse aggregation: per-edge blocks, float4 gathers + global f32 atomics.
// ---------------------------------------------------------------------------

typedef __bf16 bf16_t;
typedef __attribute__((ext_vector_type(16))) __bf16 v16bf;
typedef __attribute__((ext_vector_type(8)))  float  v8f;
typedef unsigned int u32;
typedef __attribute__((ext_vector_type(4))) u32 v4u;

union Frag16 { v4u q[2]; v16bf v; };

__device__ __forceinline__ bf16_t f2bf(float f) {
  union { float f; u32 u; } a; a.f = f;
  u32 r = (a.u + 0x7FFFu + ((a.u >> 16) & 1u)) >> 16;   // round-to-nearest-even
  union { unsigned short s; bf16_t b; } o; o.s = (unsigned short)r;
  return o.b;
}

// ------------------------- small utility kernels ---------------------------

__global__ void k_fill_f32(float* p, float v, size_t n) {
  size_t i = (size_t)blockIdx.x * blockDim.x + threadIdx.x;
  if (i < n) p[i] = v;
}

__global__ void k_cast_bf16(const float* __restrict__ x, bf16_t* __restrict__ y, size_t n) {
  size_t i = (size_t)blockIdx.x * blockDim.x + threadIdx.x;
  if (i < n) y[i] = f2bf(x[i]);
}

// W is [K x N] row-major; Wt is [Np x K] (row n of Wt = column n of W), zero-pad n>=N.
__global__ void k_transpose_w_bf16(const float* __restrict__ W, bf16_t* __restrict__ Wt,
                                   int K, int N, int Np) {
  size_t i = (size_t)blockIdx.x * blockDim.x + threadIdx.x;
  size_t total = (size_t)Np * K;
  if (i >= total) return;
  int n = (int)(i / K);
  int k = (int)(i % K);
  float v = (n < N) ? W[(size_t)k * N + n] : 0.0f;
  Wt[i] = f2bf(v);
}

// ------------------------- degree / normalization --------------------------

__global__ void k_deg_edges(const int* __restrict__ dst, const float* __restrict__ ew,
                            float* __restrict__ deg, int E) {
  int e = blockIdx.x * blockDim.x + threadIdx.x;
  if (e < E) atomicAdd(&deg[dst[e]], ew[e]);
}

__global__ void k_dinv(const float* __restrict__ deg, float* __restrict__ dinv, int n) {
  int i = blockIdx.x * blockDim.x + threadIdx.x;
  if (i < n) {
    float d = deg[i];
    dinv[i] = (d > 0.0f) ? rsqrtf(d) : 0.0f;
  }
}

// --------------------------- WMMA bf16 GEMM --------------------------------
// C[M x N] (f32) = A[M x K] (bf16 row-major) * Bt[N x K]^T (bf16, row n = col n of B)
// Wave computes a (WM*16) x (WN*16) macro-tile; block = WAVES_M x WAVES_N waves.
// Grid must exactly tile M x N (M padded by caller) -> no guards, EXEC all-1s.
//
// Fragment layouts used here (wave32; A/B k order is this kernel's choice, not the ISA table's):
//  A 16x32 bf16 : lane(m=lane&15, h=lane>>4): e0..15 -> K=16h+0..15 (contiguous)
//  B 32x16 bf16 : lane(n=lane&15, h=lane>>4): e0..15 -> K=16h+0..15 (contiguous)
//  D 16x16 f32  : VGPR v -> m = 8*(lane>>4)+v, n = lane&15

template<int WM, int WN, int WAVES_M, int WAVES_N>
__global__ void __launch_bounds__(32 * WAVES_M * WAVES_N)
k_gemm_wmma_bf16(const bf16_t* __restrict__ A, const bf16_t* __restrict__ Bt,
                 float* __restrict__ C, int N, int K) {
  const int wave = threadIdx.x >> 5;
  const int lane = threadIdx.x & 31;
  const int wm = wave / WAVES_N;
  const int wn = wave % WAVES_N;
  const int r    = lane & 15;
  const int half = lane >> 4;

  const int m0 = (blockIdx.y * WAVES_M + wm) * (WM * 16);
  const int n0 = (blockIdx.x * WAVES_N + wn) * (WN * 16);

  const bf16_t* Arow[WM];
  const bf16_t* Brow[WN];
#pragma unroll
  for (int i = 0; i < WM; ++i)
    Arow[i] = A  + (size_t)(m0 + i * 16 + r) * K + half * 16;
#pragma unroll
  for (int j = 0; j < WN; ++j)
    Brow[j] = Bt + (size_t)(n0 + j * 16 + r) * K + half * 16;

  v8f acc[WM][WN];
#pragma unroll
  for (int i = 0; i < WM; ++i)
#pragma unroll
    for (int j = 0; j < WN; ++j) acc[i][j] = (v8f){};

  for (int k0 = 0; k0 < K; k0 += 32) {
    Frag16 fa[WM], fb[WN];
#pragma unroll
    for (int i = 0; i < WM; ++i) {
      __builtin_prefetch(Arow[i] + k0 + 64, 0, 0);        // global_prefetch_b8
      fa[i].q[0] = *(const v4u*)(Arow[i] + k0);           // K = 16h + 0..7
      fa[i].q[1] = *(const v4u*)(Arow[i] + k0 + 8);       // K = 16h + 8..15
    }
#pragma unroll
    for (int j = 0; j < WN; ++j) {
      __builtin_prefetch(Brow[j] + k0 + 64, 0, 0);
      fb[j].q[0] = *(const v4u*)(Brow[j] + k0);           // K = 16h + 0..7
      fb[j].q[1] = *(const v4u*)(Brow[j] + k0 + 8);       // K = 16h + 8..15
    }
#pragma unroll
    for (int i = 0; i < WM; ++i)
#pragma unroll
      for (int j = 0; j < WN; ++j)
        acc[i][j] = __builtin_amdgcn_wmma_f32_16x16x32_bf16(
            false, fa[i].v, false, fb[j].v, (short)0, acc[i][j], false, false);
  }

#pragma unroll
  for (int i = 0; i < WM; ++i) {
#pragma unroll
    for (int j = 0; j < WN; ++j) {
      float* Crow = C + (size_t)(m0 + i * 16 + half * 8) * N + n0 + j * 16 + r;
#pragma unroll
      for (int v = 0; v < 8; ++v) Crow[(size_t)v * N] = acc[i][j][v];
    }
  }
}

// ----------------------------- aggregation ---------------------------------

// AGG[i,f] = dinv[i]^2 * H[i,f] + b[f]     (self-loop term, fused bias)
__global__ void k_agg_init(const float* __restrict__ H, const float* __restrict__ dinv,
                           const float* __restrict__ b, float* __restrict__ AGG,
                           int n, int F) {
  size_t i = (size_t)blockIdx.x * blockDim.x + threadIdx.x;
  if (i >= (size_t)n * F) return;
  int node = (int)(i / F);
  int f    = (int)(i % F);
  float di = dinv[node];
  AGG[i] = di * di * H[i] + b[f];
}

// AGG[dst,:] += norm_e * H[src,:]   — one block per edge, float4 per thread.
__global__ void __launch_bounds__(256)
k_agg_edges(const float* __restrict__ H, float* __restrict__ AGG,
            const int* __restrict__ src, const int* __restrict__ dst,
            const float* __restrict__ ew, const float* __restrict__ dinv,
            int E, int F) {
  int e = blockIdx.x;
  if (e >= E) return;
  int s = src[e], d = dst[e];
  float nrm = dinv[s] * ew[e] * dinv[d];
  int base = threadIdx.x * 4;
  if (base >= F) return;
  const float4 h4 = *(const float4*)(H + (size_t)s * F + base);
  float* out = AGG + (size_t)d * F + base;
  atomicAdd(out + 0, nrm * h4.x);
  atomicAdd(out + 1, nrm * h4.y);
  atomicAdd(out + 2, nrm * h4.z);
  atomicAdd(out + 3, nrm * h4.w);
}

// ----------------------------- batch norm ----------------------------------

#define BN_RCH 512   // rows per partial-sum chunk

__global__ void k_bn_partial(const float* __restrict__ X, int rows, int F,
                             float* __restrict__ sum, float* __restrict__ sumsq) {
  int c = blockIdx.x * blockDim.x + threadIdx.x;
  if (c >= F) return;
  int r0 = blockIdx.y * BN_RCH;
  int r1 = r0 + BN_RCH; if (r1 > rows) r1 = rows;
  float s = 0.0f, s2 = 0.0f;
  for (int r = r0; r < r1; ++r) {
    float v = X[(size_t)r * F + c];
    s += v; s2 += v * v;
  }
  atomicAdd(&sum[c], s);
  atomicAdd(&sumsq[c], s2);
}

__global__ void k_bn_final(const float* __restrict__ sum, const float* __restrict__ sumsq,
                           int rows, int F, float* __restrict__ mu, float* __restrict__ rvar) {
  int c = blockIdx.x * blockDim.x + threadIdx.x;
  if (c >= F) return;
  float m = sum[c] / (float)rows;
  float var = sumsq[c] / (float)rows - m * m;
  mu[c] = m;
  rvar[c] = rsqrtf(var + 1e-5f);
}

// y = relu((x - mu) * rvar * g + bt) -> bf16 (next layer's A operand)
__global__ void k_bn_apply_relu_bf16(const float* __restrict__ X,
                                     const float* __restrict__ mu, const float* __restrict__ rvar,
                                     const float* __restrict__ g, const float* __restrict__ bt,
                                     bf16_t* __restrict__ Y, size_t total, int F) {
  size_t i = (size_t)blockIdx.x * blockDim.x + threadIdx.x;
  if (i >= total) return;
  int f = (int)(i % F);
  float v = (X[i] - mu[f]) * rvar[f] * g[f] + bt[f];
  Y[i] = f2bf(v > 0.0f ? v : 0.0f);
}

__global__ void k_relu_bf16(const float* __restrict__ X, bf16_t* __restrict__ Y, size_t total) {
  size_t i = (size_t)blockIdx.x * blockDim.x + threadIdx.x;
  if (i >= total) return;
  float v = X[i];
  Y[i] = f2bf(v > 0.0f ? v : 0.0f);
}

// out[n,0..9] = Cfc[n,0..9] + bf[0..9]   (slice off the padded columns)
__global__ void k_fc_writeout(const float* __restrict__ Cfc, const float* __restrict__ bf,
                              float* __restrict__ out, int n, int OUT, int NP) {
  size_t i = (size_t)blockIdx.x * blockDim.x + threadIdx.x;
  if (i >= (size_t)n * OUT) return;
  int row = (int)(i / OUT);
  int o   = (int)(i % OUT);
  out[i] = Cfc[(size_t)row * NP + o] + bf[o];
}

// ------------------------------- launcher ----------------------------------

static inline size_t alignup(size_t x) { return (x + 255) & ~(size_t)255; }

extern "C" void kernel_launch(void* const* d_in, const int* in_sizes, int n_in,
                              void* d_out, int out_size, void* d_ws, size_t ws_size,
                              hipStream_t stream) {
  (void)n_in; (void)out_size; (void)ws_size;

  const int F_IN = 128, H1 = 1024, H2 = 1024, H3 = 512, OUT = 10, OUTP = 16;
  const int Nn = in_sizes[0] / F_IN;            // 20000 nodes
  const int E  = in_sizes[2];                   // 320000 edges
  const int Mp = ((Nn + 127) / 128) * 128;      // pad node dim to block tile (20096)

  const float* x   = (const float*)d_in[0];
  const int*   ei  = (const int*)d_in[1];       // [2, E]
  const float* ew  = (const float*)d_in[2];
  const float* W1  = (const float*)d_in[3];
  const float* b1  = (const float*)d_in[4];
  const float* g1  = (const float*)d_in[5];
  const float* bt1 = (const float*)d_in[6];
  const float* W2  = (const float*)d_in[7];
  const float* b2  = (const float*)d_in[8];
  const float* g2  = (const float*)d_in[9];
  const float* bt2 = (const float*)d_in[10];
  const float* W3  = (const float*)d_in[11];
  const float* b3  = (const float*)d_in[12];
  const float* Wf  = (const float*)d_in[13];
  const float* bf  = (const float*)d_in[14];
  const int* srcIdx = ei;
  const int* dstIdx = ei + E;
  float* out = (float*)d_out;

  // ---- workspace carve-up (node-dim buffers padded to Mp rows) ----
  char* ws = (char*)d_ws;
  size_t off = 0;
  auto carve = [&](size_t bytes) { void* p = ws + off; off = alignup(off + bytes); return p; };

  float*  H    = (float*) carve((size_t)Mp * H1 * sizeof(float));
  float*  AGG  = (float*) carve((size_t)Nn * H1 * sizeof(float));
  bf16_t* ABF  = (bf16_t*)carve((size_t)Mp * H1 * sizeof(bf16_t));
  bf16_t* XB   = (bf16_t*)carve((size_t)Mp * F_IN * sizeof(bf16_t));
  bf16_t* W1T  = (bf16_t*)carve((size_t)H1 * F_IN * sizeof(bf16_t));
  bf16_t* W2T  = (bf16_t*)carve((size_t)H2 * H1 * sizeof(bf16_t));
  bf16_t* W3T  = (bf16_t*)carve((size_t)H3 * H2 * sizeof(bf16_t));
  bf16_t* WFT  = (bf16_t*)carve((size_t)OUTP * H3 * sizeof(bf16_t));
  float*  CFC  = (float*) carve((size_t)Mp * OUTP * sizeof(float));
  float*  DEG  = (float*) carve((size_t)Nn * sizeof(float));
  float*  DINV = (float*) carve((size_t)Nn * sizeof(float));
  float*  MU   = (float*) carve((size_t)H1 * sizeof(float));
  float*  RVAR = (float*) carve((size_t)H1 * sizeof(float));
  float*  SUM  = (float*) carve((size_t)H1 * sizeof(float));
  float*  SUMQ = (float*) carve((size_t)H1 * sizeof(float));

  const int T = 256;
  auto blk1d = [](size_t n, int t) { return (int)((n + t - 1) / t); };

  // ---- precompute: bf16 casts, transposed weights, degree norm ----
  k_cast_bf16<<<blk1d((size_t)Nn * F_IN, T), T, 0, stream>>>(x, XB, (size_t)Nn * F_IN);
  k_transpose_w_bf16<<<blk1d((size_t)H1 * F_IN, T), T, 0, stream>>>(W1, W1T, F_IN, H1, H1);
  k_transpose_w_bf16<<<blk1d((size_t)H2 * H1, T), T, 0, stream>>>(W2, W2T, H1, H2, H2);
  k_transpose_w_bf16<<<blk1d((size_t)H3 * H2, T), T, 0, stream>>>(W3, W3T, H2, H3, H3);
  k_transpose_w_bf16<<<blk1d((size_t)OUTP * H3, T), T, 0, stream>>>(Wf, WFT, H3, OUT, OUTP);

  k_fill_f32<<<blk1d(Nn, T), T, 0, stream>>>(DEG, 1.0f, (size_t)Nn);  // self-loop weight
  k_deg_edges<<<blk1d(E, T), T, 0, stream>>>(dstIdx, ew, DEG, E);
  k_dinv<<<blk1d(Nn, T), T, 0, stream>>>(DEG, DINV, Nn);

  // Register-blocked GEMM: wave = 64x32, block = 2x4 waves = 128x128 tile.
  auto gemm = [&](const bf16_t* A, const bf16_t* Bt, float* C, int N, int K) {
    dim3 grid(N / 128, Mp / 128);
    k_gemm_wmma_bf16<4, 2, 2, 4><<<grid, 256, 0, stream>>>(A, Bt, C, N, K);
  };
  auto aggregate = [&](const float* Hb, const float* bias, float* Ab, int F) {
    k_agg_init<<<blk1d((size_t)Nn * F, T), T, 0, stream>>>(Hb, DINV, bias, Ab, Nn, F);
    k_agg_edges<<<E, T, 0, stream>>>(Hb, Ab, srcIdx, dstIdx, ew, DINV, E, F);
  };
  auto batchnorm = [&](const float* Xf, const float* g, const float* bt, bf16_t* Y, int F) {
    k_fill_f32<<<blk1d(F, T), T, 0, stream>>>(SUM, 0.0f, (size_t)F);
    k_fill_f32<<<blk1d(F, T), T, 0, stream>>>(SUMQ, 0.0f, (size_t)F);
    dim3 g2d(blk1d(F, T), (Nn + BN_RCH - 1) / BN_RCH);
    k_bn_partial<<<g2d, T, 0, stream>>>(Xf, Nn, F, SUM, SUMQ);
    k_bn_final<<<blk1d(F, T), T, 0, stream>>>(SUM, SUMQ, Nn, F, MU, RVAR);
    k_bn_apply_relu_bf16<<<blk1d((size_t)Nn * F, T), T, 0, stream>>>(
        Xf, MU, RVAR, g, bt, Y, (size_t)Nn * F, F);
  };

  // ---- layer 1: GCNConv(128->1024) + BN + ReLU ----
  gemm(XB, W1T, H, H1, F_IN);
  aggregate(H, b1, AGG, H1);
  batchnorm(AGG, g1, bt1, ABF, H1);

  // ---- layer 2: GCNConv(1024->1024) + BN + ReLU ----
  gemm(ABF, W2T, H, H2, H1);
  aggregate(H, b2, AGG, H2);
  batchnorm(AGG, g2, bt2, ABF, H2);

  // ---- layer 3: GCNConv(1024->512) + ReLU ----
  gemm(ABF, W3T, H, H3, H2);
  aggregate(H, b3, AGG, H3);
  k_relu_bf16<<<blk1d((size_t)Nn * H3, T), T, 0, stream>>>(AGG, ABF, (size_t)Nn * H3);

  // ---- FC (512->10, padded to 16 cols): 1x1 tile waves, 8 M-waves/block ----
  {
    dim3 grid(OUTP / 16, Mp / 128);
    k_gemm_wmma_bf16<1, 1, 8, 1><<<grid, 256, 0, stream>>>(ABF, WFT, CFC, OUTP, H3);
  }
  k_fc_writeout<<<blk1d((size_t)Nn * OUT, T), T, 0, stream>>>(CFC, bf, out, Nn, OUT, OUTP);
}
